// MMDAlignmentLayer_47382079210039
// MI455X (gfx1250) — matrix operand order repaired and checked by reading, not compiled, not run on hardware
//
#include <hip/hip_runtime.h>

#define DIMK 256        // feature dim
#define NROW 8192       // total rows (source + target)
#define NS   4096       // source rows
#define BT   128        // macro tile edge
#define KC   32         // K chunk staged in LDS (= one bf16 WMMA K-step)
#define SA   40         // padded LDS row stride in bf16 elems (KC + 8)
#define NBLK 64         // macro tiles per dimension (8192/128)

// workspace layout
#define SQ_OFF     1024                   // float offset: sq[8192]
#define PARTL_OFF  9216                   // float offset: per-block partials [4096]
#define SPART_OFF  16384                  // float offset: column-sum partials [64][256]
#define HPLANE_B   131072                 // byte offset: bf16 hi plane [8192*256]
#define LPLANE_B   (HPLANE_B + NROW * DIMK * 2)  // bf16 lo plane

typedef __attribute__((ext_vector_type(16))) __bf16 v16bf;
typedef __attribute__((ext_vector_type(8)))  __bf16 v8bf;
typedef __attribute__((ext_vector_type(8)))  float  v8f;

__device__ __forceinline__ const float* mmd_row_ptr(const float* s, const float* t, int row) {
  return (row < NS) ? (s + (size_t)row * DIMK) : (t + (size_t)(row - NS) * DIMK);
}

// ---- split fp32 -> bf16 hi/lo planes (one pass, stays hot in L2) ------------
__global__ void mmd_cvt(const float* __restrict__ src, const float* __restrict__ tgt,
                        __bf16* __restrict__ hp, __bf16* __restrict__ lp) {
  int idx = (blockIdx.x * 256 + threadIdx.x) * 4;      // element in concat matrix
  int row = idx >> 8;
  const float* rp = mmd_row_ptr(src, tgt, row);
  float4 x = *(const float4*)(rp + (idx & 255));
  __bf16 h0 = (__bf16)x.x, h1 = (__bf16)x.y, h2 = (__bf16)x.z, h3 = (__bf16)x.w;
  hp[idx + 0] = h0; hp[idx + 1] = h1; hp[idx + 2] = h2; hp[idx + 3] = h3;
  lp[idx + 0] = (__bf16)(x.x - (float)h0);
  lp[idx + 1] = (__bf16)(x.y - (float)h1);
  lp[idx + 2] = (__bf16)(x.z - (float)h2);
  lp[idx + 3] = (__bf16)(x.w - (float)h3);
}

// ---- per-row squared norms: one wave per row --------------------------------
__global__ void mmd_row_sq(const float* __restrict__ src, const float* __restrict__ tgt,
                           float* __restrict__ sq) {
  int lane = threadIdx.x & 31;
  int wave = threadIdx.x >> 5;
  int row  = blockIdx.x * 8 + wave;
  const float4* rp = (const float4*)mmd_row_ptr(src, tgt, row);
  float4 a = rp[lane * 2 + 0];
  float4 b = rp[lane * 2 + 1];
  float acc = a.x*a.x + a.y*a.y + a.z*a.z + a.w*a.w
            + b.x*b.x + b.y*b.y + b.z*b.z + b.w*b.w;
  for (int o = 16; o > 0; o >>= 1) acc += __shfl_xor(acc, o, 32);
  if (lane == 0) sq[row] = acc;
}

// ---- column-sum partials (for ||sum_i x_i||^2) ------------------------------
__global__ void mmd_col_part(const float* __restrict__ src, const float* __restrict__ tgt,
                             float* __restrict__ s_part) {
  int t = threadIdx.x;
  int b = blockIdx.x;
  const float* base = mmd_row_ptr(src, tgt, b * 128);
  float acc = 0.f;
  for (int r = 0; r < 128; ++r) acc += base[(size_t)r * DIMK + t];
  s_part[b * 256 + t] = acc;
}

// ---- closed-form bandwidth:  sum(d2) = 2n*sum(sq) - 2*||colsum||^2 ----------
__global__ void mmd_bw(const float* __restrict__ sq, const float* __restrict__ s_part,
                       float* __restrict__ invbw_out) {
  __shared__ float r1[256], r2[256];
  int t = threadIdx.x;
  float a1 = 0.f;
  for (int k = 0; k < NROW / 256; ++k) a1 += sq[t + 256 * k];
  float sv = 0.f;
  for (int b = 0; b < 64; ++b) sv += s_part[b * 256 + t];
  r1[t] = a1; r2[t] = sv * sv;
  __syncthreads();
  for (int s2 = 128; s2 > 0; s2 >>= 1) {
    if (t < s2) { r1[t] += r1[t + s2]; r2[t] += r2[t + s2]; }
    __syncthreads();
  }
  if (t == 0) {
    double sumsq = (double)r1[0], sn2 = (double)r2[0];
    double tot = 2.0 * (double)NROW * sumsq - 2.0 * sn2;
    double bw  = tot / ((double)NROW * (double)NROW - (double)NROW) / 4.0; // / MUL^(NUM//2)
    invbw_out[0] = (float)(1.0 / bw);
  }
}

__global__ void mmd_zero(float* __restrict__ p) {
  p[blockIdx.x * 256 + threadIdx.x] = 0.f;
}

// ---- fused bf16x3 WMMA GEMM -> d2 -> 5x exp -> signed block partial ---------
__global__ void __launch_bounds__(256) mmd_main(
    const __bf16* __restrict__ hp, const __bf16* __restrict__ lp,
    const float* __restrict__ sq, const float* __restrict__ invbwp,
    float* __restrict__ partials) {
  int bi = blockIdx.y, bj = blockIdx.x;
  if (bj < bi) return;                       // symmetric: upper triangle only

  __shared__ __bf16 Ah[BT * SA];
  __shared__ __bf16 Al[BT * SA];
  __shared__ __bf16 Bh[BT * SA];
  __shared__ __bf16 Bl[BT * SA];
  __shared__ float  red[256];

  int t    = threadIdx.x;
  int lane = t & 31, wave = t >> 5;
  int lrow = lane & 15, lhi = lane >> 4;
  int i0 = bi * BT, j0 = bj * BT;

  v8f acc[8] = {};

  // staging decomposition: thread -> (row, 16-elem segment) of the 128x32 chunk
  int srow = t >> 1;
  int sseg = (t & 1) * 16;
  int lds_s = srow * SA + sseg;

  // A frag: lanes 0-15 hold K 0-15, lanes 16-31 hold K 16-31 (contiguous)
  int aoff = (wave * 16 + lrow) * SA + 16 * lhi;

  for (int kc = 0; kc < DIMK; kc += KC) {
    size_t ago = (size_t)(i0 + srow) * DIMK + kc + sseg;
    size_t bgo = (size_t)(j0 + srow) * DIMK + kc + sseg;
    *(v8bf*)&Ah[lds_s] = *(const v8bf*)(hp + ago);
    *(v8bf*)&Ah[lds_s + 8] = *(const v8bf*)(hp + ago + 8);
    *(v8bf*)&Al[lds_s] = *(const v8bf*)(lp + ago);
    *(v8bf*)&Al[lds_s + 8] = *(const v8bf*)(lp + ago + 8);
    *(v8bf*)&Bh[lds_s] = *(const v8bf*)(hp + bgo);
    *(v8bf*)&Bh[lds_s + 8] = *(const v8bf*)(hp + bgo + 8);
    *(v8bf*)&Bl[lds_s] = *(const v8bf*)(lp + bgo);
    *(v8bf*)&Bl[lds_s + 8] = *(const v8bf*)(lp + bgo + 8);
    if (kc + KC < DIMK) {
      __builtin_prefetch(hp + ago + KC, 0, 1);
      __builtin_prefetch(lp + ago + KC, 0, 1);
      __builtin_prefetch(hp + bgo + KC, 0, 1);
      __builtin_prefetch(lp + bgo + KC, 0, 1);
    }
    __syncthreads();

    v8bf ah0 = *(const v8bf*)&Ah[aoff];
    v8bf ah1 = *(const v8bf*)&Ah[aoff + 8];
    v8bf al0 = *(const v8bf*)&Al[aoff];
    v8bf al1 = *(const v8bf*)&Al[aoff + 8];
    v16bf a_hi = __builtin_shufflevector(ah0, ah1, 0,1,2,3,4,5,6,7,8,9,10,11,12,13,14,15);
    v16bf a_lo = __builtin_shufflevector(al0, al1, 0,1,2,3,4,5,6,7,8,9,10,11,12,13,14,15);

#pragma unroll
    for (int tj = 0; tj < 8; ++tj) {
      // B frag: lanes 0-15 hold K 0-15, lanes 16-31 hold K 16-31 (contiguous)
      int boff = (tj * 16 + lrow) * SA + 16 * lhi;
      v8bf bh0 = *(const v8bf*)&Bh[boff];
      v8bf bh1 = *(const v8bf*)&Bh[boff + 8];
      v8bf bl0 = *(const v8bf*)&Bl[boff];
      v8bf bl1 = *(const v8bf*)&Bl[boff + 8];
      v16bf b_hi = __builtin_shufflevector(bh0, bh1, 0,1,2,3,4,5,6,7,8,9,10,11,12,13,14,15);
      v16bf b_lo = __builtin_shufflevector(bl0, bl1, 0,1,2,3,4,5,6,7,8,9,10,11,12,13,14,15);
      // dot = hi*hi + hi*lo + lo*hi   (lo*lo term ~2^-16 relative: dropped)
      acc[tj] = __builtin_amdgcn_wmma_f32_16x16x32_bf16(
          false, a_hi, false, b_hi, (short)0, acc[tj], false, false);
      acc[tj] = __builtin_amdgcn_wmma_f32_16x16x32_bf16(
          false, a_hi, false, b_lo, (short)0, acc[tj], false, false);
      acc[tj] = __builtin_amdgcn_wmma_f32_16x16x32_bf16(
          false, a_lo, false, b_hi, (short)0, acc[tj], false, false);
    }
    __syncthreads();
  }

  // epilogue: d2 = sq_i + sq_j - 2*dot, clamp, 5-bandwidth Gaussian sum
  float invbw = invbwp[0];
  float sqi[8];
#pragma unroll
  for (int r = 0; r < 8; ++r) sqi[r] = sq[i0 + wave * 16 + lhi * 8 + r];

  float local = 0.f;
#pragma unroll
  for (int tj = 0; tj < 8; ++tj) {
    float sqj = sq[j0 + tj * 16 + lrow];
#pragma unroll
    for (int r = 0; r < 8; ++r) {
      float d2 = fmaxf(sqi[r] + sqj - 2.0f * acc[tj][r], 0.0f);
      float u  = -d2 * invbw;
      local += __expf(u) + __expf(0.5f * u) + __expf(0.25f * u)
             + __expf(0.125f * u) + __expf(0.0625f * u);
    }
  }

  red[t] = local;
  __syncthreads();
  for (int s2 = 128; s2 > 0; s2 >>= 1) {
    if (t < s2) red[t] += red[t + s2];
    __syncthreads();
  }
  if (t == 0) {
    float sgn  = ((bi < NBLK / 2) == (bj < NBLK / 2)) ? 1.0f : -1.0f;
    float mult = (bi == bj) ? 1.0f : 2.0f;   // off-diagonal tiles mirrored
    partials[bi * NBLK + bj] = red[0] * sgn * mult * (1.0f / ((float)NS * (float)NS));
  }
}

// ---- final deterministic reduction + clamp ----------------------------------
__global__ void mmd_final(const float* __restrict__ partials, float* __restrict__ out) {
  __shared__ float red[256];
  int t = threadIdx.x;
  float a = 0.f;
  for (int k = 0; k < 16; ++k) a += partials[t + 256 * k];
  red[t] = a;
  __syncthreads();
  for (int s2 = 128; s2 > 0; s2 >>= 1) {
    if (t < s2) red[t] += red[t + s2];
    __syncthreads();
  }
  if (t == 0) out[0] = fmaxf(red[0], 0.0f);
}

extern "C" void kernel_launch(void* const* d_in, const int* in_sizes, int n_in,
                              void* d_out, int out_size, void* d_ws, size_t ws_size,
                              hipStream_t stream) {
  const float* src = (const float*)d_in[0];
  const float* tgt = (const float*)d_in[1];
  float* ws  = (float*)d_ws;
  float* out = (float*)d_out;
  __bf16* hp = (__bf16*)((char*)d_ws + HPLANE_B);
  __bf16* lp = (__bf16*)((char*)d_ws + LPLANE_B);

  mmd_zero<<<NBLK * NBLK / 256, 256, 0, stream>>>(ws + PARTL_OFF);
  mmd_cvt<<<NROW * DIMK / 4 / 256, 256, 0, stream>>>(src, tgt, hp, lp);
  mmd_row_sq<<<NROW / 8, 256, 0, stream>>>(src, tgt, ws + SQ_OFF);
  mmd_col_part<<<64, 256, 0, stream>>>(src, tgt, ws + SPART_OFF);
  mmd_bw<<<1, 256, 0, stream>>>(ws + SQ_OFF, ws + SPART_OFF, ws);
  mmd_main<<<dim3(NBLK, NBLK), 256, 0, stream>>>(hp, lp, ws + SQ_OFF, ws, ws + PARTL_OFF);
  mmd_final<<<1, 256, 0, stream>>>(ws + PARTL_OFF, out);
}
